// Net_2740189135512
// MI455X (gfx1250) — matrix operand order repaired and checked by reading, not compiled, not run on hardware
//
#include <hip/hip_runtime.h>

typedef _Float16 h16;
typedef __attribute__((ext_vector_type(8)))  _Float16 v8h;
typedef __attribute__((ext_vector_type(16))) _Float16 v16h;
typedef __attribute__((ext_vector_type(8)))  float    v8f;

#define BB 2
#define C4 256
#define C0 64
#define HH 160
#define WW 320
#define HW (HH*WW)          // 51200
#define BH (BB*HH)          // 320
#define NX ((size_t)BB*C4*HW)   // 26,214,400
#define NQ ((size_t)BB*C0*HW)   // 6,553,600
#define NSC ((size_t)BH*WW*WW)  // 32,768,000
#define GROUPS 4
#define EPSBN 1e-5f
#define PSTR 72   // padded ic-stride in LDS patch (144 B: 16B-aligned, low bank conflicts)

#define CAT16(lo, hi) __builtin_shufflevector((lo), (hi), 0,1,2,3,4,5,6,7,8,9,10,11,12,13,14,15)

// ---------------- fragment loaders (CDNA5 16-bit WMMA layouts) ----------------
// A 16x32 f16: lane row = lane%16; element e: K = 16*half + e
__device__ __forceinline__ v16h load_a16(const h16* __restrict__ p, int hf) {
  v8h lo = *(const v8h*)(p + 16*hf);
  v8h hi = *(const v8h*)(p + 16*hf + 8);
  return CAT16(lo, hi);
}
// B 32x16 f16 from B^T storage: lane col = lane%16; element e: K = 16*half + e
__device__ __forceinline__ v16h load_b16(const h16* __restrict__ p, int hf) {
  v8h lo = *(const v8h*)(p + 16*hf);
  v8h hi = *(const v8h*)(p + 16*hf + 8);
  return CAT16(lo, hi);
}
#define WMMA_F16(a, b, c) \
  __builtin_amdgcn_wmma_f32_16x16x32_f16(false, (a), false, (b), (short)0, (c), false, false)

// ---------------- BN statistics ----------------
__global__ void bn_stats_k(const float* __restrict__ catL, const float* __restrict__ catR,
                           const float* __restrict__ gamma, const float* __restrict__ beta,
                           float* __restrict__ scale, float* __restrict__ shift) {
  int c = blockIdx.x, side = blockIdx.y;
  const float* x = side ? catR : catL;
  __shared__ float s1[256], s2[256];
  float sum = 0.f, sq = 0.f;
  for (int n = 0; n < BB; n++) {
    const float* p = x + ((size_t)n*C4 + c)*HW;
    for (int i = threadIdx.x; i < HW; i += blockDim.x) { float v = p[i]; sum += v; sq += v*v; }
  }
  s1[threadIdx.x] = sum; s2[threadIdx.x] = sq; __syncthreads();
  for (int st = 128; st > 0; st >>= 1) {
    if (threadIdx.x < st) { s1[threadIdx.x] += s1[threadIdx.x+st]; s2[threadIdx.x] += s2[threadIdx.x+st]; }
    __syncthreads();
  }
  if (threadIdx.x == 0) {
    float N = (float)(BB*HW);
    float mu = s1[0]/N, var = s2[0]/N - mu*mu;
    float sc = gamma[c] * rsqrtf(var + EPSBN);
    scale[side*C4 + c] = sc;
    shift[side*C4 + c] = beta[c] - mu*sc;
  }
}

__global__ void bn_apply_k(const float* __restrict__ x, const float* __restrict__ scale,
                           const float* __restrict__ shift, h16* __restrict__ out) {
  size_t i = (size_t)blockIdx.x*blockDim.x + threadIdx.x;
  if (i >= NX) return;
  int c = (int)((i / HW) % C4);
  out[i] = (h16)(x[i]*scale[c] + shift[c]);
}

// repack 3x3 grouped conv weights (256,64,3,3) f32 -> f16 [oc][(kh*3+kw)*64+ic]
__global__ void wcvt_k(const float* __restrict__ w, h16* __restrict__ out) {
  int i = blockIdx.x*blockDim.x + threadIdx.x;
  if (i >= C4*576) return;
  int oc = i/576, k = i%576;
  int ic = k%64, khw = k/64, kh = khw/3, kw = khw%3;
  out[i] = (h16)w[((oc*64 + ic)*3 + kh)*3 + kw];
}

__global__ void cvt16_k(const float* __restrict__ x, h16* __restrict__ y, size_t n) {
  size_t i = (size_t)blockIdx.x*blockDim.x + threadIdx.x;
  if (i < n) y[i] = (h16)x[i];
}

// ---------------- grouped 3x3 conv, implicit GEMM with WMMA ----------------
// block 256 = 8 waves; tile: 64 oc x 32 px for one (b, g, row hh)
// LDS patch layout [kh][x][ic] (ic contiguous, stride PSTR) so B fragments are
// two 16B ds_load_b128 instead of 16 scalar u16 loads.
template<bool LEAKY, bool RESID>
__global__ void conv3x3_k(const h16* __restrict__ Xin, const h16* __restrict__ wh,
                          const float* __restrict__ bias, const h16* __restrict__ resid,
                          h16* __restrict__ out) {
  __shared__ __align__(16) h16 patch[3*34*PSTR];
  int b = blockIdx.z / GROUPS, g = blockIdx.z % GROUPS;
  int hh = blockIdx.y, w0 = blockIdx.x*32;
  int tid = threadIdx.x;
  for (int idx = tid; idx < 64*3*34; idx += 256) {
    int ic = idx/102, rem = idx%102, r = rem/34, x = rem%34;   // coalesced global reads
    int y = hh + r - 1, xx = w0 + x - 1;
    h16 v = (h16)0.f;
    if (y >= 0 && y < HH && xx >= 0 && xx < WW)
      v = Xin[(((size_t)b*C4 + g*64 + ic)*HH + y)*WW + xx];
    patch[(r*34 + x)*PSTR + ic] = v;
  }
  __syncthreads();
  int wave = tid >> 5, lane = tid & 31, hf = lane >> 4, l16 = lane & 15;
  int ocT = wave >> 1, pxT = wave & 1;
  int ocRow = g*64 + ocT*16 + l16;   // A-fragment row (weight row)
  int pxb = pxT*16;
  v8f acc = {};
  for (int s = 0; s < 18; s++) {     // K = 576, steps of 32: (kh,kw) x icHalf
    int khw = s >> 1, icHalf = s & 1;
    int kh = khw/3, kw = khw%3;
    const h16* wp = wh + (size_t)ocRow*576 + khw*64 + icHalf*32;
    v16h a = load_a16(wp, hf);
    // B element e: ic = icHalf*32 + 16*hf + e at fixed (kh, x=pxb+l16+kw) -> contiguous
    const h16* bpp = &patch[(kh*34 + pxb + l16 + kw)*PSTR + icHalf*32 + 16*hf];
    v8h blo = *(const v8h*)bpp;
    v8h bhi = *(const v8h*)(bpp + 8);
    v16h bf = CAT16(blo, bhi);
    acc = WMMA_F16(a, bf, acc);
  }
  int wout = w0 + pxb + l16;         // D col = lane%16
#pragma unroll
  for (int r = 0; r < 8; r++) {      // D row = r + 8*half
    int occ = g*64 + ocT*16 + r + 8*hf;
    float v = acc[r] + bias[occ];
    if (LEAKY) v = (v >= 0.f) ? v : 0.1f*v;
    size_t oidx = (((size_t)b*C4 + occ)*HH + hh)*WW + wout;
    if (RESID) v += (float)resid[oidx];
    out[oidx] = (h16)v;
  }
}

// ---------------- 1x1 grouped projection ----------------
__global__ void proj_k(const h16* __restrict__ feat, const float* __restrict__ wq,
                       const float* __restrict__ bq, float* __restrict__ out) {
  size_t i = (size_t)blockIdx.x*blockDim.x + threadIdx.x;
  if (i >= NQ) return;
  int w = (int)(i % WW); size_t t = i / WW;
  int h = (int)(t % HH); t /= HH;
  int oc = (int)(t % C0); int b = (int)(t / C0);
  int g = oc >> 4;
  const float* wr = wq + oc*64;
  const h16* fp = feat + (((size_t)b*C4 + g*64)*HH + h)*WW + w;
  float acc = bq[oc];
#pragma unroll 8
  for (int ic = 0; ic < 64; ic++) acc += wr[ic] * (float)fp[(size_t)ic*HW];
  out[i] = acc;
}

// subtract per-(b,c,h) width-mean, pack to (b*h, w, 64) f16
__global__ void meansub_k(const float* __restrict__ Qraw, h16* __restrict__ Qws) {
  int c = blockIdx.x, h = blockIdx.y, b = blockIdx.z;
  __shared__ float sm[WW];
  const float* p = Qraw + (((size_t)b*C0 + c)*HH + h)*WW;
  int tid = threadIdx.x;
  float v = p[tid];
  sm[tid] = v; __syncthreads();
  if (tid < 160) sm[tid] += sm[tid+160]; __syncthreads();
  if (tid < 80)  sm[tid] += sm[tid+80];  __syncthreads();
  if (tid < 40)  sm[tid] += sm[tid+40];  __syncthreads();
  if (tid < 20)  sm[tid] += sm[tid+20];  __syncthreads();
  if (tid < 10)  sm[tid] += sm[tid+10];  __syncthreads();
  if (tid < 5)   sm[tid] += sm[tid+5];   __syncthreads();
  float m = (sm[0]+sm[1]+sm[2]+sm[3]+sm[4]) * (1.f/WW);
  Qws[(((size_t)(b*HH + h))*WW + tid)*C0 + c] = (h16)(v - m);
}

// ---------------- score GEMM: 320x64x320 per (b,h), WMMA ----------------
__global__ void score_k(const h16* __restrict__ Qws, const h16* __restrict__ Kws,
                        float* __restrict__ score) {
  int u0 = blockIdx.x*16, v0 = blockIdx.y*16, bh = blockIdx.z;
  int lane = threadIdx.x, hf = lane >> 4, l16 = lane & 15;
  const h16* ap = Qws + ((size_t)bh*WW + u0 + l16)*C0;
  const h16* bp = Kws + ((size_t)bh*WW + v0 + l16)*C0;   // K^T rows are contiguous
  v8f acc = {};
#pragma unroll
  for (int k0 = 0; k0 < 64; k0 += 32)
    acc = WMMA_F16(load_a16(ap + k0, hf), load_b16(bp + k0, hf), acc);
  float* op = score + (size_t)bh*WW*WW;
#pragma unroll
  for (int r = 0; r < 8; r++) {
    int u = u0 + r + 8*hf;
    op[(size_t)u*WW + v0 + l16] = acc[r];
  }
}

// ---------------- dual softmax ----------------
__device__ __forceinline__ float red320(float v, float* sm, int tid, bool ismax) {
  sm[tid] = v; __syncthreads();
  if (tid < 160) sm[tid] = ismax ? fmaxf(sm[tid], sm[tid+160]) : sm[tid]+sm[tid+160]; __syncthreads();
  if (tid < 80)  sm[tid] = ismax ? fmaxf(sm[tid], sm[tid+80])  : sm[tid]+sm[tid+80];  __syncthreads();
  if (tid < 40)  sm[tid] = ismax ? fmaxf(sm[tid], sm[tid+40])  : sm[tid]+sm[tid+40];  __syncthreads();
  if (tid < 20)  sm[tid] = ismax ? fmaxf(sm[tid], sm[tid+20])  : sm[tid]+sm[tid+20];  __syncthreads();
  if (tid < 10)  sm[tid] = ismax ? fmaxf(sm[tid], sm[tid+10])  : sm[tid]+sm[tid+10];  __syncthreads();
  if (tid < 5)   sm[tid] = ismax ? fmaxf(sm[tid], sm[tid+5])   : sm[tid]+sm[tid+5];   __syncthreads();
  float r = ismax ? fmaxf(fmaxf(fmaxf(sm[0],sm[1]),fmaxf(sm[2],sm[3])),sm[4])
                  : (sm[0]+sm[1]+sm[2]+sm[3]+sm[4]);
  __syncthreads();
  return r;
}

__global__ void smrow_k(const float* __restrict__ score, h16* __restrict__ M) {
  size_t row = blockIdx.x;                  // bh*W + u
  __shared__ float sm[WW];
  int tid = threadIdx.x;
  float v = score[row*WW + tid];
  float mx = red320(v, sm, tid, true);
  float e = expf(v - mx);
  float tot = red320(e, sm, tid, false);
  M[row*WW + tid] = (h16)(e / tot);
}

__global__ void smcol_k(const float* __restrict__ score, h16* __restrict__ M) {
  size_t col = blockIdx.x;                  // bh*W + v
  size_t bh = col / WW; int v = (int)(col % WW);
  __shared__ float sm[WW];
  int tid = threadIdx.x;
  float x = score[(bh*WW + (size_t)tid)*WW + v];
  float mx = red320(x, sm, tid, true);
  float e = expf(x - mx);
  float tot = red320(e, sm, tid, false);
  M[col*WW + tid] = (h16)(e / tot);         // M_ltr[bh, v, u]
}

// ---------------- V = einsum(relax(Ma)[i,:], Mb[:,i]) ----------------
__global__ void vrelax_k(const h16* __restrict__ Ma, const h16* __restrict__ Mb,
                         float* __restrict__ V) {
  int i = (int)(blockIdx.x % WW); size_t bh = blockIdx.x / WW;
  const h16* A = Ma + bh*WW*WW;
  const h16* Bm = Mb + bh*WW*WW;
  __shared__ float sm[128];
  float acc = 0.f;
  for (int j = threadIdx.x; j < WW; j += 128) {
    float s = 0.f;
#pragma unroll
    for (int k = -2; k <= 2; k++) {
      int ii = i + k;
      if (ii >= 0 && ii < WW) s += (float)A[(size_t)ii*WW + j];
    }
    acc += s * (float)Bm[(size_t)j*WW + i];
  }
  sm[threadIdx.x] = acc; __syncthreads();
  for (int st = 64; st > 0; st >>= 1) {
    if (threadIdx.x < st) sm[threadIdx.x] += sm[threadIdx.x+st];
    __syncthreads();
  }
  if (threadIdx.x == 0) V[blockIdx.x] = sm[0];
}

// ---------------- warping GEMM: xT[bh,u,c] = sum_v M[bh,u,v]*xh[b,c,h,v] ----------------
__global__ void warp_k(const h16* __restrict__ M, const h16* __restrict__ xh,
                       float* __restrict__ xT) {
  int c0 = blockIdx.x*16, u0 = blockIdx.y*16, bh = blockIdx.z;
  int b = bh / HH, h = bh % HH;
  int lane = threadIdx.x, hf = lane >> 4, l16 = lane & 15;
  const h16* ap = M + ((size_t)bh*WW + u0 + l16)*WW;
  const h16* bp = xh + (((size_t)b*C0 + c0 + l16)*HH + h)*WW;  // B^T: contiguous in v
  v8f acc = {};
#pragma unroll
  for (int k0 = 0; k0 < WW; k0 += 32)
    acc = WMMA_F16(load_a16(ap + k0, hf), load_b16(bp + k0, hf), acc);
#pragma unroll
  for (int r = 0; r < 8; r++) {
    int u = u0 + r + 8*hf;
    xT[((size_t)bh*WW + u)*C0 + c0 + l16] = acc[r];
  }
}

// ---------------- final tanh-gated blend ----------------
__global__ void blend_k(const float* __restrict__ x, const float* __restrict__ xT,
                        const float* __restrict__ V, float* __restrict__ out) {
  size_t i = (size_t)blockIdx.x*blockDim.x + threadIdx.x;
  if (i >= NQ) return;
  int w = (int)(i % WW); size_t t = i / WW;
  int h = (int)(t % HH); t /= HH;
  int c = (int)(t % C0); int b = (int)(t / C0);
  size_t bh = (size_t)b*HH + h;
  float g = tanhf(5.f * V[bh*WW + w]);
  out[i] = x[i]*(1.f - g) + xT[(bh*WW + w)*C0 + c]*g;
}

extern "C" void kernel_launch(void* const* d_in, const int* in_sizes, int n_in,
                              void* d_out, int out_size, void* d_ws, size_t ws_size,
                              hipStream_t stream) {
  (void)in_sizes; (void)n_in; (void)out_size; (void)ws_size;
  const float* x_left  = (const float*)d_in[0];
  const float* x_right = (const float*)d_in[1];
  const float* catL    = (const float*)d_in[2];
  const float* catR    = (const float*)d_in[3];
  const float* bq_w    = (const float*)d_in[4];
  const float* bq_b    = (const float*)d_in[5];
  const float* bs_w    = (const float*)d_in[6];
  const float* bs_b    = (const float*)d_in[7];
  const float* rb_w1   = (const float*)d_in[8];
  const float* rb_b1   = (const float*)d_in[9];
  const float* rb_w2   = (const float*)d_in[10];
  const float* rb_b2   = (const float*)d_in[11];
  const float* gamma   = (const float*)d_in[12];
  const float* beta    = (const float*)d_in[13];
  float* out = (float*)d_out;

  char* ws = (char*)d_ws;
  size_t off = 0;
  auto take = [&](size_t bytes) -> char* {
    char* p = ws + off;
    off += (bytes + 255) & ~(size_t)255;
    return p;
  };
  float* bnscale = (float*)take(2*C4*4);
  float* bnshift = (float*)take(2*C4*4);
  h16*   wh1     = (h16*)take((size_t)C4*576*2);
  h16*   wh2     = (h16*)take((size_t)C4*576*2);
  h16*   XhL     = (h16*)take(NX*2);
  h16*   XhR     = (h16*)take(NX*2);
  h16*   y1      = (h16*)take(NX*2);
  h16*   feat    = (h16*)take(NX*2);
  float* Qraw    = (float*)take(NQ*4);
  float* Kraw    = (float*)take(NQ*4);
  h16*   Qws     = (h16*)take(NQ*2);
  h16*   Kws     = (h16*)take(NQ*2);
  h16*   xlh     = (h16*)take(NQ*2);
  h16*   xrh     = (h16*)take(NQ*2);
  float* score   = (float*)take(NSC*4);
  h16*   Mrtl    = (h16*)take(NSC*2);
  h16*   Mltr    = (h16*)take(NSC*2);
  float* Vl      = (float*)take((size_t)BH*WW*4);
  float* Vr      = (float*)take((size_t)BH*WW*4);
  float* xLT     = (float*)take(NQ*4);
  float* xRT     = (float*)take(NQ*4);

  // 1) BN stats + apply (both sides), weight/input conversions
  bn_stats_k<<<dim3(C4,2), 256, 0, stream>>>(catL, catR, gamma, beta, bnscale, bnshift);
  {
    size_t gr = (NX + 255)/256;
    bn_apply_k<<<(unsigned)gr, 256, 0, stream>>>(catL, bnscale,     bnshift,     XhL);
    bn_apply_k<<<(unsigned)gr, 256, 0, stream>>>(catR, bnscale+C4,  bnshift+C4,  XhR);
  }
  wcvt_k<<<(C4*576+255)/256, 256, 0, stream>>>(rb_w1, wh1);
  wcvt_k<<<(C4*576+255)/256, 256, 0, stream>>>(rb_w2, wh2);
  cvt16_k<<<(unsigned)((NQ+255)/256), 256, 0, stream>>>(x_left,  xlh, NQ);
  cvt16_k<<<(unsigned)((NQ+255)/256), 256, 0, stream>>>(x_right, xrh, NQ);

  // 2) ResB (conv-lrelu-conv-residual) + 1x1 proj + mean-sub, per side (reuse y1/feat)
  dim3 cg(WW/32, HH, BB*GROUPS);
  conv3x3_k<true,false><<<cg, 256, 0, stream>>>(XhL, wh1, rb_b1, nullptr, y1);
  conv3x3_k<false,true><<<cg, 256, 0, stream>>>(y1,  wh2, rb_b2, XhL,     feat);
  proj_k<<<(unsigned)((NQ+255)/256), 256, 0, stream>>>(feat, bq_w, bq_b, Qraw);
  meansub_k<<<dim3(C0, HH, BB), WW, 0, stream>>>(Qraw, Qws);

  conv3x3_k<true,false><<<cg, 256, 0, stream>>>(XhR, wh1, rb_b1, nullptr, y1);
  conv3x3_k<false,true><<<cg, 256, 0, stream>>>(y1,  wh2, rb_b2, XhR,     feat);
  proj_k<<<(unsigned)((NQ+255)/256), 256, 0, stream>>>(feat, bs_w, bs_b, Kraw);
  meansub_k<<<dim3(C0, HH, BB), WW, 0, stream>>>(Kraw, Kws);

  // 3) attention
  score_k<<<dim3(WW/16, WW/16, BH), 32, 0, stream>>>(Qws, Kws, score);
  smrow_k<<<(unsigned)((size_t)BH*WW), WW, 0, stream>>>(score, Mrtl);
  smcol_k<<<(unsigned)((size_t)BH*WW), WW, 0, stream>>>(score, Mltr);
  vrelax_k<<<(unsigned)((size_t)BH*WW), 128, 0, stream>>>(Mrtl, Mltr, Vl);
  vrelax_k<<<(unsigned)((size_t)BH*WW), 128, 0, stream>>>(Mltr, Mrtl, Vr);

  // 4) warp + blend
  warp_k<<<dim3(C0/16, WW/16, BH), 32, 0, stream>>>(Mrtl, xrh, xLT);
  warp_k<<<dim3(C0/16, WW/16, BH), 32, 0, stream>>>(Mltr, xlh, xRT);
  blend_k<<<(unsigned)((NQ+255)/256), 256, 0, stream>>>(x_left,  xLT, Vl, out);
  blend_k<<<(unsigned)((NQ+255)/256), 256, 0, stream>>>(x_right, xRT, Vr, out + NQ);
}
